// FERMI_75402445849094
// MI455X (gfx1250) — matrix operand order repaired and checked by reading, not compiled, not run on hardware
//
#include <hip/hip_runtime.h>

// Problem constants (from reference)
#define B_N  131072
#define D_K  784
#define H_N  100
#define NT   7              // ceil(100/16) N-tiles (padded H = 112)
#define NCH  25             // ceil(784/32) K-chunks of 32 (padded K = 800)
#define LAM  0.1f
#define WAVES 8
#define ROWS_PER_BLOCK (WAVES * 16)          // 128
#define NBLK (B_N / ROWS_PER_BLOCK)          // 1024

// d_ws layout (bytes):
//   [0,               8192)               : block partials (2 * NBLK floats)
//   [8192,            8192+179200)        : Bhi  (NT*NCH*32 lanes * 16 bf16)
//   [8192+179200,     8192+2*179200)      : Blo
#define WS_PART_OFF 0
#define WS_BHI_OFF  8192
#define WS_B_BYTES  (NT * NCH * 32 * 16 * 2)     // 179200
#define WS_BLO_OFF  (WS_BHI_OFF + WS_B_BYTES)

typedef __attribute__((ext_vector_type(4)))  float  v4f;
typedef __attribute__((ext_vector_type(8)))  float  v8f;
typedef __attribute__((ext_vector_type(16))) __bf16 v16bf;

// ---------------------------------------------------------------------------
// Prep: split fc1_w into bf16 hi/lo, packed in the exact WMMA B-operand
// layout. B 32x16 bf16 layout: lane = grp*16 + N; lane holds K = 8*grp + e for
// e < 8, K = 16 + 8*grp + (e-8) for e >= 8. Pads n in [100,112) and k in [784,800) with zeros.
// One element per thread; total NT*NCH*32*16 = 89600 elements.
// ---------------------------------------------------------------------------
__global__ __launch_bounds__(256) void fermi_prep_kernel(
    const float* __restrict__ fc1w, __bf16* __restrict__ Bhi,
    __bf16* __restrict__ Blo)
{
    int gid = blockIdx.x * blockDim.x + threadIdx.x;
    if (gid >= NT * NCH * 32 * 16) return;
    int e    = gid & 15;
    int lane = (gid >> 4) & 31;
    int tc   = gid >> 9;
    int c    = tc % NCH;
    int t    = tc / NCH;
    int n    = t * 16 + (lane & 15);
    int k    = c * 32 + (e < 8 ? (lane >> 4) * 8 + e : 16 + (lane >> 4) * 8 + (e - 8));
    float v  = (n < H_N && k < D_K) ? fc1w[n * D_K + k] : 0.0f;
    __bf16 h = (__bf16)v;
    Bhi[gid] = h;
    Blo[gid] = (__bf16)(v - (float)h);
}

// Convert 4 f32 -> bf16 hi/lo into vector slots [base, base+4)
__device__ __forceinline__ void cvt4(const v4f f, v16bf& hi, v16bf& lo, int base)
{
#pragma unroll
    for (int j = 0; j < 4; ++j) {
        float v = f[j];
        __bf16 h = (__bf16)v;
        hi[base + j] = h;
        lo[base + j] = (__bf16)(v - (float)h);
    }
}

// 7 N-tiles, 3 split-products each: Ahi*Bhi + Alo*Bhi + Ahi*Blo
__device__ __forceinline__ void mm_tiles(v8f* acc, v16bf ahi, v16bf alo,
                                         const __bf16* __restrict__ Bhi,
                                         const __bf16* __restrict__ Blo,
                                         int c, int lane)
{
#pragma unroll
    for (int t = 0; t < NT; ++t) {
        size_t off = ((size_t)(t * NCH + c) * 32 + lane) * 16;
        v16bf bh = *(const v16bf*)(Bhi + off);
        v16bf bl = *(const v16bf*)(Blo + off);
        acc[t] = __builtin_amdgcn_wmma_f32_16x16x32_bf16(
            false, ahi, false, bh, (short)0, acc[t], false, false);
        acc[t] = __builtin_amdgcn_wmma_f32_16x16x32_bf16(
            false, alo, false, bh, (short)0, acc[t], false, false);
        acc[t] = __builtin_amdgcn_wmma_f32_16x16x32_bf16(
            false, ahi, false, bl, (short)0, acc[t], false, false);
    }
}

// ---------------------------------------------------------------------------
// Main: split-bf16 GEMM (X @ fc1_w.T) + fused relu/fc2/sigmoid epilogue,
// per-block partial sums of yh^2 and yh * (S @ (P@W)).
// ---------------------------------------------------------------------------
__global__ __launch_bounds__(256) void fermi_gemm_kernel(
    const float* __restrict__ X,    const float* __restrict__ S,
    const float* __restrict__ W,    const float* __restrict__ fc1b,
    const float* __restrict__ fc2w, const float* __restrict__ fc2b,
    const float* __restrict__ P,    const __bf16* __restrict__ Bhi,
    const __bf16* __restrict__ Blo, float* __restrict__ partials)
{
    const int tid  = threadIdx.x;
    const int wave = tid >> 5;
    const int lane = tid & 31;
    const int grp  = lane >> 4;     // lane-group: 0 (lanes 0-15) / 1 (16-31)
    const int lid  = lane & 15;
    const int rowBase = (blockIdx.x * WAVES + wave) * 16;

    // A operand (16x32 bf16): lane holds row M=lid,
    // elements e<8 -> K = kb + 8*grp + e ; e>=8 -> K = kb + 16 + 8*grp + (e-8)
    const float* rowPtr = X + (size_t)(rowBase + lid) * D_K;

    v8f acc[NT];
#pragma unroll
    for (int t = 0; t < NT; ++t)
        acc[t] = (v8f){0.f, 0.f, 0.f, 0.f, 0.f, 0.f, 0.f, 0.f};

    // Full 32-K chunks: 24 of them (K = 0..767)
#pragma unroll 2
    for (int c = 0; c < NCH - 1; ++c) {
        const int kb = c * 32;
        v4f f0 = *(const v4f*)(rowPtr + kb + 8 * grp);
        v4f f1 = *(const v4f*)(rowPtr + kb + 8 * grp + 4);
        v4f f2 = *(const v4f*)(rowPtr + kb + 16 + 8 * grp);
        v4f f3 = *(const v4f*)(rowPtr + kb + 16 + 8 * grp + 4);
        v16bf ahi, alo;
        cvt4(f0, ahi, alo, 0);
        cvt4(f1, ahi, alo, 4);
        cvt4(f2, ahi, alo, 8);
        cvt4(f3, ahi, alo, 12);
        mm_tiles(acc, ahi, alo, Bhi, Blo, c, lane);
    }

    // Tail chunk c = 24: K = 768..783 valid, 784..799 zero-padded in B.
    {
        const int c  = NCH - 1;
        const int kb = c * 32;
        v4f f0 = *(const v4f*)(rowPtr + kb + 8 * grp);
        v4f f1 = *(const v4f*)(rowPtr + kb + 8 * grp + 4);
        v16bf ahi, alo;
        cvt4(f0, ahi, alo, 0);
        cvt4(f1, ahi, alo, 4);
#pragma unroll
        for (int e = 8; e < 16; ++e) { ahi[e] = (__bf16)0.f; alo[e] = (__bf16)0.f; }
        mm_tiles(acc, ahi, alo, Bhi, Blo, c, lane);
    }

    // Epilogue. C/D layout: VGPR r, lane -> (M = r + 8*grp, N = lid)
    float part[8];
#pragma unroll
    for (int r = 0; r < 8; ++r) part[r] = 0.f;

#pragma unroll
    for (int t = 0; t < NT; ++t) {
        int n = t * 16 + lid;
        float bias = (n < H_N) ? fc1b[n] : 0.f;
        float w2   = (n < H_N) ? fc2w[n] : 0.f;   // zero-masks padded cols
#pragma unroll
        for (int r = 0; r < 8; ++r) {
            float hv = acc[t][r] + bias;
            hv = fmaxf(hv, 0.f);
            part[r] = fmaf(hv, w2, part[r]);
        }
    }

    // Reduce over the 16 lanes of each lane-group (masks < 16 stay in-group)
#pragma unroll
    for (int r = 0; r < 8; ++r) {
#pragma unroll
        for (int off = 1; off < 16; off <<= 1)
            part[r] += __shfl_xor(part[r], off, 32);
    }

    __shared__ float red[WAVES][2][2];
    if (lid == 0) {
        float w0 = W[0], w1 = W[1];
        float pw0 = P[0] * w0 + P[1] * w1;   // (P @ W)[0]
        float pw1 = P[2] * w0 + P[3] * w1;   // (P @ W)[1]
        float b2  = fc2b[0];
        float s_yh2 = 0.f, s_ysw = 0.f;
#pragma unroll
        for (int r = 0; r < 8; ++r) {
            int row  = rowBase + grp * 8 + r;
            float z  = part[r] + b2;
            float yh = 1.0f / (1.0f + expf(-z));
            float sPW = S[2 * row] * pw0 + S[2 * row + 1] * pw1;
            s_yh2 += yh * yh;
            s_ysw += yh * sPW;
        }
        red[wave][grp][0] = s_yh2;
        red[wave][grp][1] = s_ysw;
    }
    __syncthreads();

    if (tid == 0) {
        float a0 = 0.f, a1 = 0.f;
        for (int w = 0; w < WAVES; ++w)
            for (int g = 0; g < 2; ++g) { a0 += red[w][g][0]; a1 += red[w][g][1]; }
        partials[blockIdx.x * 2 + 0] = a0;
        partials[blockIdx.x * 2 + 1] = a1;
    }
}

// ---------------------------------------------------------------------------
// Final reduction: sum 1024 block partials, apply the scalar formula.
// ---------------------------------------------------------------------------
__global__ __launch_bounds__(256) void fermi_final_kernel(
    const float* __restrict__ partials, const float* __restrict__ W,
    float* __restrict__ out)
{
    __shared__ float s0[256], s1[256];
    float a0 = 0.f, a1 = 0.f;
    for (int i = threadIdx.x; i < NBLK; i += 256) {
        a0 += partials[2 * i + 0];
        a1 += partials[2 * i + 1];
    }
    s0[threadIdx.x] = a0;
    s1[threadIdx.x] = a1;
    __syncthreads();
    for (int off = 128; off > 0; off >>= 1) {
        if (threadIdx.x < off) {
            s0[threadIdx.x] += s0[threadIdx.x + off];
            s1[threadIdx.x] += s1[threadIdx.x + off];
        }
        __syncthreads();
    }
    if (threadIdx.x == 0) {
        float trWWt = W[0] * W[0] + W[1] * W[1];
        float n = (float)B_N;
        float summation = -s0[0] * trWWt + 2.0f * s1[0] - n;
        out[0] = LAM * summation / n;
    }
}

// ---------------------------------------------------------------------------
extern "C" void kernel_launch(void* const* d_in, const int* in_sizes, int n_in,
                              void* d_out, int out_size, void* d_ws, size_t ws_size,
                              hipStream_t stream) {
    const float* X    = (const float*)d_in[0];
    const float* S    = (const float*)d_in[1];
    const float* W    = (const float*)d_in[2];
    const float* fc1w = (const float*)d_in[3];
    const float* fc1b = (const float*)d_in[4];
    const float* fc2w = (const float*)d_in[5];
    const float* fc2b = (const float*)d_in[6];
    const float* P    = (const float*)d_in[7];

    float*  partials = (float*)((char*)d_ws + WS_PART_OFF);
    __bf16* Bhi      = (__bf16*)((char*)d_ws + WS_BHI_OFF);
    __bf16* Blo      = (__bf16*)((char*)d_ws + WS_BLO_OFF);

    const int prepElems = NT * NCH * 32 * 16;            // 89600
    fermi_prep_kernel<<<(prepElems + 255) / 256, 256, 0, stream>>>(fc1w, Bhi, Blo);
    fermi_gemm_kernel<<<NBLK, 256, 0, stream>>>(X, S, W, fc1b, fc2w, fc2b, P,
                                                Bhi, Blo, partials);
    fermi_final_kernel<<<1, 256, 0, stream>>>(partials, W, (float*)d_out);
}
